// RandConvBlock_54331336294842
// MI455X (gfx1250) — compile-verified, matrix operand order checked by reading, not run on hardware
//
#include <hip/hip_runtime.h>
#include <hip/hip_bf16.h>
#include <math.h>

typedef _Float16 half8  __attribute__((ext_vector_type(8)));
typedef _Float16 half16 __attribute__((ext_vector_type(16)));
typedef float    f32x8  __attribute__((ext_vector_type(8)));

#define NB    32
#define CIN   64
#define COUT  64
#define HH    64
#define WW    64
#define K2    9
#define KDIM  (CIN * K2)       // 576
#define HW    (HH * WW)        // 4096
#define BROW  584              // padded LDS row stride (halves); 584*2 = 1168 B = 73*16

// ---------------------------------------------------------------------------
// k0: convert weight f32 -> f16.  weight flat [O][C][3][3] == [O][c*9+k].
// ---------------------------------------------------------------------------
__global__ __launch_bounds__(256)
void wcvt_kernel(const float* __restrict__ w, _Float16* __restrict__ wh) {
    int i = blockIdx.x * 256 + threadIdx.x;
    if (i < COUT * KDIM) wh[i] = (_Float16)w[i];
}

// ---------------------------------------------------------------------------
// k1: fused bilinear-sample + WMMA GEMM.
// One workgroup = one (batch, output row ho): M=64 outputs x N=64 pixels,
// K=576 reduction.  8 waves, each owns 2 M-tiles x 1 N-tile (16x16 each).
// ---------------------------------------------------------------------------
__global__ __launch_bounds__(256)
void dconv_kernel(const float* __restrict__ x, const float* __restrict__ off,
                  const _Float16* __restrict__ wh, float* __restrict__ d) {
    extern __shared__ _Float16 Bt[];               // [64 pixels][BROW]  (B^T, f16)
    const int blk = blockIdx.x;
    const int b   = blk >> 6;
    const int ho  = blk & 63;
    const int tid = threadIdx.x;

    const float* xb = x   + (size_t)b * CIN * HW;
    const float* ob = off + (size_t)b * 2 * K2 * HW + (size_t)ho * WW;

    // ---- Phase A: bilinear gather -> LDS  Bt[wo][c*9+kk] ----
    for (int pair = tid; pair < K2 * WW; pair += 256) {
        const int kk = pair >> 6;                  // tap 0..8
        const int wo = pair & 63;                  // pixel (lane-contiguous)
        const float dy = ob[(2 * kk)     * HW + wo];
        const float dx = ob[(2 * kk + 1) * HW + wo];
        const float py = (float)(ho - 1 + kk / 3) + dy;
        const float px = (float)(wo - 1 + kk % 3) + dx;
        const float y0f = floorf(py), x0f = floorf(px);
        const float fy = py - y0f,   fx = px - x0f;
        const int y0 = (int)y0f, x0 = (int)x0f;
        const int y1 = y0 + 1,  x1 = x0 + 1;
        const float vy0 = (y0 >= 0 && y0 < HH) ? 1.f : 0.f;
        const float vy1 = (y1 >= 0 && y1 < HH) ? 1.f : 0.f;
        const float vx0 = (x0 >= 0 && x0 < WW) ? 1.f : 0.f;
        const float vx1 = (x1 >= 0 && x1 < WW) ? 1.f : 0.f;
        const int cy0 = min(max(y0, 0), HH - 1), cy1 = min(max(y1, 0), HH - 1);
        const int cx0 = min(max(x0, 0), WW - 1), cx1 = min(max(x1, 0), WW - 1);
        const int i00 = cy0 * WW + cx0, i01 = cy0 * WW + cx1;
        const int i10 = cy1 * WW + cx0, i11 = cy1 * WW + cx1;
        const float w00 = (1.f - fy) * (1.f - fx) * vy0 * vx0;
        const float w01 = (1.f - fy) * fx         * vy0 * vx1;
        const float w10 = fy * (1.f - fx)         * vy1 * vx0;
        const float w11 = fy * fx                 * vy1 * vx1;
        _Float16* dst = &Bt[wo * BROW + kk];
        const float* p = xb;
        #pragma unroll 4
        for (int c = 0; c < CIN; ++c) {
            float v = w00 * p[i00] + w01 * p[i01] + w10 * p[i10] + w11 * p[i11];
            dst[c * K2] = (_Float16)v;
            p += HW;
        }
    }
    __syncthreads();

    // ---- Phase B: WMMA GEMM  d[o][wo] = sum_k wh[o][k] * Bt[wo][k] ----
    const int wave  = tid >> 5;
    const int lane  = tid & 31;
    const int ntile = wave & 3;                    // N tile: 16 pixels
    const int mbase = (wave >> 2) * 32;            // 2 M tiles: mbase, mbase+16
    const int lhalf = lane >> 4;                   // fragment half select
    const int lidx  = lane & 15;

    union { half16 v; half8 h[2]; } a0, a1, bb;
    f32x8 acc0 = {}; f32x8 acc1 = {};

    const _Float16* Bp = &Bt[(ntile * 16 + lidx) * BROW + lhalf * 8];
    const _Float16* A0 = wh + (size_t)(mbase + lidx) * KDIM + lhalf * 8;
    const _Float16* A1 = A0 + (size_t)16 * KDIM;

    for (int kc = 0; kc < KDIM; kc += 32) {        // 18 iterations
        a0.h[0] = *(const half8*)(A0 + kc);
        a0.h[1] = *(const half8*)(A0 + kc + 16);
        a1.h[0] = *(const half8*)(A1 + kc);
        a1.h[1] = *(const half8*)(A1 + kc + 16);
        bb.h[0] = *(const half8*)(Bp + kc);        // two ds_load_b128
        bb.h[1] = *(const half8*)(Bp + kc + 16);
        acc0 = __builtin_amdgcn_wmma_f32_16x16x32_f16(
                   false, a0.v, false, bb.v, (short)0, acc0, false, false);
        acc1 = __builtin_amdgcn_wmma_f32_16x16x32_f16(
                   false, a1.v, false, bb.v, (short)0, acc1, false, false);
    }

    // C/D layout: VGPR r, lanes 0-15 -> M=r, lanes 16-31 -> M=r+8; N = lane&15
    float* dbase = d + (size_t)b * COUT * HW + (size_t)ho * WW;
    const int wo = ntile * 16 + lidx;
    #pragma unroll
    for (int r = 0; r < 8; ++r) {
        const int o0 = mbase + r + lhalf * 8;
        dbase[(size_t)o0 * HW + wo]        = acc0[r];
        dbase[(size_t)(o0 + 16) * HW + wo] = acc1[r];
    }
}

// ---------------------------------------------------------------------------
// k2: per-(b,o) instance norm (unbiased std) + tanh.
// One workgroup per (b,o); 4096 floats held in registers (16/thread).
// ---------------------------------------------------------------------------
__global__ __launch_bounds__(256)
void norm_tanh_kernel(const float* __restrict__ d, const float* __restrict__ gamma,
                      const float* __restrict__ beta, float* __restrict__ out) {
    __shared__ float s1[256];
    __shared__ float s2[256];
    __shared__ float mb[2];
    const int bo  = blockIdx.x;
    const int tid = threadIdx.x;
    const float* p = d   + (size_t)bo * HW;
    float*       q = out + (size_t)bo * HW;

    float4 v[4];
    float sum = 0.f, sq = 0.f;
    #pragma unroll
    for (int i = 0; i < 4; ++i) {
        v[i] = ((const float4*)p)[tid + 256 * i];
        sum += v[i].x + v[i].y + v[i].z + v[i].w;
        sq  += v[i].x * v[i].x + v[i].y * v[i].y + v[i].z * v[i].z + v[i].w * v[i].w;
    }
    s1[tid] = sum; s2[tid] = sq;
    __syncthreads();
    for (int st = 128; st > 0; st >>= 1) {
        if (tid < st) { s1[tid] += s1[tid + st]; s2[tid] += s2[tid + st]; }
        __syncthreads();
    }
    if (tid == 0) {
        const float m   = s1[0] * (1.f / 4096.f);
        const float var = (s2[0] - 4096.f * m * m) * (1.f / 4095.f);   // ddof=1
        mb[0] = m;
        mb[1] = sqrtf(fmaxf(var, 0.f));
    }
    __syncthreads();
    const float m   = mb[0];
    const float inv = gamma[0] / (mb[1] + 1e-8f);
    const float bt  = beta[0];
    #pragma unroll
    for (int i = 0; i < 4; ++i) {
        float4 r;
        r.x = tanhf((v[i].x - m) * inv + bt);
        r.y = tanhf((v[i].y - m) * inv + bt);
        r.z = tanhf((v[i].z - m) * inv + bt);
        r.w = tanhf((v[i].w - m) * inv + bt);
        ((float4*)q)[tid + 256 * i] = r;
    }
}

// ---------------------------------------------------------------------------
extern "C" void kernel_launch(void* const* d_in, const int* in_sizes, int n_in,
                              void* d_out, int out_size, void* d_ws, size_t ws_size,
                              hipStream_t stream) {
    const float* x     = (const float*)d_in[0];   // [32,64,64,64]
    const float* off   = (const float*)d_in[1];   // [32,18,64,64]
    const float* w     = (const float*)d_in[2];   // [64,64,3,3]
    const float* gamma = (const float*)d_in[3];   // scalar
    const float* beta  = (const float*)d_in[4];   // scalar
    float*       out   = (float*)d_out;           // [32,64,64,64]

    _Float16* wh  = (_Float16*)d_ws;                          // 73,728 B (f16 weights)
    float*    dbf = (float*)((char*)d_ws + 73728);            // 33.5 MB  (conv result)

    wcvt_kernel<<<(COUT * KDIM + 255) / 256, 256, 0, stream>>>(w, wh);

    const size_t lds = (size_t)WW * BROW * sizeof(_Float16); // 74,752 B
    dconv_kernel<<<NB * HH, 256, lds, stream>>>(x, off, wh, dbf);

    norm_tanh_kernel<<<NB * COUT, 256, 0, stream>>>(dbf, gamma, beta, out);
}
